// SAGE_Re_80041010528552
// MI455X (gfx1250) — matrix operand order repaired and checked by reading, not compiled, not run on hardware
//
#include <hip/hip_runtime.h>

// ---------------------------------------------------------------------------
// Types / feature probes for CDNA5
// ---------------------------------------------------------------------------
typedef __attribute__((ext_vector_type(16))) __bf16        v16bf;
typedef __attribute__((ext_vector_type(8)))  float         v8f;
typedef __attribute__((ext_vector_type(8)))  unsigned int  v8u;
typedef __attribute__((ext_vector_type(4)))  int           v4i;

#if __has_builtin(__builtin_amdgcn_global_load_async_to_lds_b128) && \
    __has_builtin(__builtin_amdgcn_s_wait_asynccnt)
#define HAVE_ASYNC 1
#else
#define HAVE_ASYNC 0
#endif

// ---------------------------------------------------------------------------
// f32 -> bf16 round-to-nearest-even, packed pair: 2x(bfe+add3) + 1 v_perm
// ---------------------------------------------------------------------------
__device__ __forceinline__ unsigned int f2bf_pk(float a, float b) {
    unsigned int ua = __builtin_bit_cast(unsigned int, a);
    unsigned int ub = __builtin_bit_cast(unsigned int, b);
    ua += 0x7fffu + ((ua >> 16) & 1u);
    ub += 0x7fffu + ((ub >> 16) & 1u);
#if __has_builtin(__builtin_amdgcn_perm)
    // D = [ua.b2, ua.b3, ub.b2, ub.b3] : lo16 = bf16(a), hi16 = bf16(b)
    return __builtin_amdgcn_perm(ub, ua, 0x07060302u);
#else
    return (ua >> 16) | (ub & 0xffff0000u);
#endif
}

__device__ __forceinline__ unsigned short f2bf(float f) {
    unsigned int u = __builtin_bit_cast(unsigned int, f);
    u += 0x7fffu + ((u >> 16) & 1u);
    return (unsigned short)(u >> 16);
}

#if HAVE_ASYNC
typedef __attribute__((address_space(1))) v4i* as1_v4i;
typedef __attribute__((address_space(3))) v4i* as3_v4i;
__device__ __forceinline__ void async_copy16(const void* g, void* l) {
    __builtin_amdgcn_global_load_async_to_lds_b128(
        (as1_v4i)(v4i*)const_cast<void*>(g), (as3_v4i)(v4i*)l, 0, 0);
}
__device__ __forceinline__ void async_wait0() {
    __builtin_amdgcn_s_wait_asynccnt(0);
}
#endif

// ---------------------------------------------------------------------------
// Kernel 1a (fallback path): per-row stats only
// ---------------------------------------------------------------------------
__global__ void stats_kernel(const float* __restrict__ adj,
                             float* __restrict__ dis,
                             float* __restrict__ scale,
                             int n) {
    const int row = blockIdx.x;
    const int tid = threadIdx.x;
    const float* a = adj + (long long)row * n;
    float deg = 0.0f;
    int nnz = 0;
    for (int j = tid; j < n; j += blockDim.x) {
        float v = a[j];
        deg += v;
        nnz += (v > 0.0f) ? 1 : 0;
    }
    __shared__ float sd[256];
    __shared__ int   sn[256];
    sd[tid] = deg; sn[tid] = nnz;
    __syncthreads();
    for (int s = 128; s > 0; s >>= 1) {
        if (tid < s) { sd[tid] += sd[tid + s]; sn[tid] += sn[tid + s]; }
        __syncthreads();
    }
    if (tid == 0) {
        float d = rsqrtf(sd[0]);
        dis[row] = d;
        float nn = (sn[0] > 0) ? (float)sn[0] : 1.0f;
        scale[row] = d / nn;
    }
}

// ---------------------------------------------------------------------------
// Kernel 1b (fast path): fused stats + f32->bf16 adjacency convert with
// zero padding to [Mp x Kp]. One block per padded row.
// ---------------------------------------------------------------------------
__global__ void stats_convert_kernel(const float* __restrict__ adj,
                                     unsigned short* __restrict__ adjB,
                                     float* __restrict__ dis,
                                     float* __restrict__ scale,
                                     int n, int Kp) {
    const int row = blockIdx.x;
    const int tid = threadIdx.x;
    unsigned int* rowB = (unsigned int*)(adjB + (size_t)row * Kp);
    const int half = Kp >> 1;
    if (row >= n) {                      // zero-pad rows
        for (int idx = tid; idx < half; idx += blockDim.x) rowB[idx] = 0u;
        return;
    }
    const float* a = adj + (long long)row * n;
    float deg = 0.0f;
    int nnz = 0;
    for (int idx = tid; idx < half; idx += blockDim.x) {
        int k = idx << 1;
        float v0 = (k     < n) ? a[k]     : 0.0f;
        float v1 = (k + 1 < n) ? a[k + 1] : 0.0f;
        deg += v0 + v1;
        nnz += ((v0 > 0.0f) ? 1 : 0) + ((v1 > 0.0f) ? 1 : 0);
        rowB[idx] = f2bf_pk(v0, v1);     // exact for 0/1
    }
    __shared__ float sd[256];
    __shared__ int   sn[256];
    sd[tid] = deg; sn[tid] = nnz;
    __syncthreads();
    for (int s = 128; s > 0; s >>= 1) {
        if (tid < s) { sd[tid] += sd[tid + s]; sn[tid] += sn[tid + s]; }
        __syncthreads();
    }
    if (tid == 0) {
        float d = rsqrtf(sd[0]);
        dis[row] = d;
        float nn = (sn[0] > 0) ? (float)sn[0] : 1.0f;
        scale[row] = d / nn;
    }
}

// ---------------------------------------------------------------------------
// Common tiling parameters (64x64 tile, K-step 32, 8 waves, wave tile 16x32)
// ---------------------------------------------------------------------------
#define BM 64
#define BN 64
#define BK 32
#define LDSB_S 34   // B LDS stride (ushort): 68B rows, u32-aligned
#define LDSA_S 40   // A LDS stride for async path: 80B rows, 16B-aligned

// ---------------------------------------------------------------------------
// Kernel 2: general f32-operand bf16-WMMA GEMM (flexible epilogue).
// grid.x = N tiles, grid.y = M tiles  (A rows shared in L2 across grid.x)
// ---------------------------------------------------------------------------
__launch_bounds__(256)
__global__ void gemm_bf16_kernel(const float* __restrict__ A, long long lda,
                                 const float* __restrict__ B, long long ldb,
                                 float* __restrict__ C, long long ldc,
                                 int M, int Ngl, int K,
                                 const float* __restrict__ bRowScale,
                                 const float* __restrict__ rowscale,
                                 const float* __restrict__ alphaPtr,
                                 const float* __restrict__ residual, long long ldr,
                                 const float* __restrict__ cadd, long long ldadd,
                                 const float* __restrict__ bias,
                                 int relu) {
    __shared__ unsigned short Alds[BM][LDSB_S];
    __shared__ unsigned short Blds[BN][LDSB_S];

    const int tid  = threadIdx.x;
    const int lane = tid & 31;
    const int wid  = tid >> 5;
    const int l15  = lane & 15;
    const int lhi  = lane >> 4;
    const int wm   = wid & 3;
    const int wn2  = wid >> 2;

    const long long rowBase = (long long)blockIdx.y * BM;
    const long long colBase = (long long)blockIdx.x * BN;

    const int sr  = tid >> 2;
    const int sk0 = (tid & 3) << 3;
    const int bk  = tid >> 3;
    const int bn0 = (tid & 7) << 3;

    const long long gArow = rowBase + sr;
    const long long gBcol = colBase + bn0;

    const bool mFull = (rowBase + BM) <= (long long)M;
    const bool nFull = (colBase + BN) <= (long long)Ngl;

    const long long aRowC = (gArow < M) ? gArow : (long long)(M - 1);
    const float aMask = (gArow < M) ? 1.0f : 0.0f;

    v8f acc0 = {};
    v8f acc1 = {};
    float aR[8], bR[8];

    auto loadTiles = [&](int k0, float* aOut, float* bOut) {
        const bool kFull = (k0 + BK) <= K;
        if (mFull && kFull) {
            const float4* p = (const float4*)(A + gArow * lda + (k0 + sk0));
            float4 u = p[0], v = p[1];
            aOut[0] = u.x; aOut[1] = u.y; aOut[2] = u.z; aOut[3] = u.w;
            aOut[4] = v.x; aOut[5] = v.y; aOut[6] = v.z; aOut[7] = v.w;
        } else {
            const float* ar = A + aRowC * lda;
#pragma unroll
            for (int j = 0; j < 8; ++j) {
                int c = k0 + sk0 + j;
                int cc = (c < K) ? c : (K - 1);
                float cm = (c < K) ? aMask : 0.0f;
                aOut[j] = ar[cc] * cm;
            }
        }
        {
            int gk = k0 + bk;
            int gkc = (gk < K) ? gk : (K - 1);
            float s = bRowScale ? bRowScale[gkc] : 1.0f;
            if (nFull && kFull) {
                const float4* p = (const float4*)(B + (long long)gk * ldb + gBcol);
                float4 u = p[0], v = p[1];
                bOut[0] = u.x * s; bOut[1] = u.y * s; bOut[2] = u.z * s; bOut[3] = u.w * s;
                bOut[4] = v.x * s; bOut[5] = v.y * s; bOut[6] = v.z * s; bOut[7] = v.w * s;
            } else {
                float km = (gk < K) ? 1.0f : 0.0f;
                const float* br = B + (long long)gkc * ldb;
#pragma unroll
                for (int j = 0; j < 8; ++j) {
                    long long n = gBcol + j;
                    long long nc = (n < Ngl) ? n : (long long)(Ngl - 1);
                    float nm = (n < Ngl) ? km : 0.0f;
                    bOut[j] = br[nc] * s * nm;
                }
            }
        }
    };

    loadTiles(0, aR, bR);

    int k0 = 0;
    while (true) {
        __syncthreads();
        {
            unsigned int* ad = (unsigned int*)&Alds[sr][sk0];
#pragma unroll
            for (int j = 0; j < 4; ++j) ad[j] = f2bf_pk(aR[2 * j], aR[2 * j + 1]);
#pragma unroll
            for (int j = 0; j < 4; ++j) {
                unsigned int pk = f2bf_pk(bR[2 * j], bR[2 * j + 1]);
                Blds[bn0 + 2 * j][bk]     = (unsigned short)pk;          // ds_store_b16
                Blds[bn0 + 2 * j + 1][bk] = (unsigned short)(pk >> 16);  // ds_store_b16_d16_hi
            }
        }
        __syncthreads();

        const int kn = k0 + BK;
        const bool more = kn < K;
        float aN[8], bN[8];
        if (more) loadTiles(kn, aN, bN);

        v8u au;
        {
            const unsigned int* arow = (const unsigned int*)(&Alds[wm * 16 + l15][0]);
#pragma unroll
            for (int v = 0; v < 8; ++v) {
                int K0 = ((v >> 2) << 4) + (lhi << 3) + ((v & 3) << 1);
                au[v] = arow[K0 >> 1];
            }
        }
        v16bf afrag = __builtin_bit_cast(v16bf, au);

        v8u bu0, bu1;
        {
            const unsigned int* b0r = (const unsigned int*)(&Blds[wn2 * 32 + l15][0]);
            const unsigned int* b1r = (const unsigned int*)(&Blds[wn2 * 32 + 16 + l15][0]);
#pragma unroll
            for (int v = 0; v < 8; ++v) {
                int idx = ((v >> 2) << 3) + (lhi << 2) + (v & 3);
                bu0[v] = b0r[idx];
                bu1[v] = b1r[idx];
            }
        }
        v16bf bfrag0 = __builtin_bit_cast(v16bf, bu0);
        v16bf bfrag1 = __builtin_bit_cast(v16bf, bu1);

        acc0 = __builtin_amdgcn_wmma_f32_16x16x32_bf16(
            false, afrag, false, bfrag0, (short)0, acc0, false, false);
        acc1 = __builtin_amdgcn_wmma_f32_16x16x32_bf16(
            false, afrag, false, bfrag1, (short)0, acc1, false, false);

        if (!more) break;
#pragma unroll
        for (int j = 0; j < 8; ++j) { aR[j] = aN[j]; bR[j] = bN[j]; }
        k0 = kn;
    }

    const float alpha = alphaPtr ? alphaPtr[0] : 1.0f;
#pragma unroll
    for (int j = 0; j < 2; ++j) {
        v8f acc = j ? acc1 : acc0;
        long long n = colBase + wn2 * 32 + j * 16 + l15;
#pragma unroll
        for (int r = 0; r < 8; ++r) {
            long long m = rowBase + wm * 16 + lhi * 8 + r;
            if (m < M && n < Ngl) {
                float val = acc[r];
                if (rowscale) val *= rowscale[m];
                val *= alpha;
                if (cadd)     val += cadd[m * ldadd + n];
                if (bias)     val += bias[n];
                if (residual) val += residual[m * ldr + n];
                if (relu)     val = fmaxf(val, 0.0f);
                C[m * ldc + n] = val;
            }
        }
    }
}

// ---------------------------------------------------------------------------
// Kernel 3: aggregation GEMM with pre-converted, zero-padded bf16 A.
//   C[m][n] = rowscale[m] * sum_k A[m][k] * bf16(bRowScale[k]*B[k][n])
// A tiles staged with GLOBAL_LOAD_ASYNC_TO_LDS_B128 (double-buffered LDS),
// falling back to uint4-through-registers when the builtin is unavailable.
// ---------------------------------------------------------------------------
__launch_bounds__(256)
__global__ void gemm_aggB_kernel(const unsigned short* __restrict__ A, long long lda,
                                 const float* __restrict__ B, long long ldb,
                                 float* __restrict__ C, long long ldc,
                                 int M, int Ngl, int K,
                                 const float* __restrict__ bRowScale,
                                 const float* __restrict__ rowscale) {
    __shared__ unsigned short Alds[2][BM][LDSA_S];
    __shared__ unsigned short Blds[2][BN][LDSB_S];

    const int tid  = threadIdx.x;
    const int lane = tid & 31;
    const int wid  = tid >> 5;
    const int l15  = lane & 15;
    const int lhi  = lane >> 4;
    const int wm   = wid & 3;
    const int wn2  = wid >> 2;

    const long long rowBase = (long long)blockIdx.y * BM;
    const long long colBase = (long long)blockIdx.x * BN;

    const int sr  = tid >> 2;           // A: row in tile, 16B chunk at sk0
    const int sk0 = (tid & 3) << 3;
    const int bk  = tid >> 3;           // B: k row, 8 n's at bn0
    const int bn0 = (tid & 7) << 3;

    const long long gArow = rowBase + sr;     // always valid (padded)
    const long long gBcol = colBase + bn0;

    const bool nFull = (colBase + BN) <= (long long)Ngl;
    const int  Kp    = (K + BK - 1) & ~(BK - 1);  // padded A width (lda >= Kp)

    const unsigned short* aSrc = A + gArow * lda + sk0;

#if !HAVE_ASYNC
    uint4 aC;
#endif
    auto issueA = [&](int k0, int pbuf) {
#if HAVE_ASYNC
        async_copy16(aSrc + k0, &Alds[pbuf][sr][sk0]);
#else
        (void)pbuf;
        aC = *(const uint4*)(aSrc + k0);
#endif
    };
    auto landA = [&](int pbuf) {
#if HAVE_ASYNC
        (void)pbuf;
        async_wait0();                 // own 16B slice of current buffer arrived
#else
        *(uint4*)&Alds[pbuf][sr][sk0] = aC;
#endif
    };

    auto loadB = [&](int k0, float* bOut) {
        const bool kFull = (k0 + BK) <= K;
        int gk = k0 + bk;
        int gkc = (gk < K) ? gk : (K - 1);
        float s = bRowScale ? bRowScale[gkc] : 1.0f;
        if (nFull && kFull) {
            const float4* p = (const float4*)(B + (long long)gk * ldb + gBcol);
            float4 u = p[0], v = p[1];
            bOut[0] = u.x * s; bOut[1] = u.y * s; bOut[2] = u.z * s; bOut[3] = u.w * s;
            bOut[4] = v.x * s; bOut[5] = v.y * s; bOut[6] = v.z * s; bOut[7] = v.w * s;
        } else {
            float km = (gk < K) ? 1.0f : 0.0f;
            const float* br = B + (long long)gkc * ldb;
#pragma unroll
            for (int j = 0; j < 8; ++j) {
                long long n = gBcol + j;
                long long nc = (n < Ngl) ? n : (long long)(Ngl - 1);
                float nm = (n < Ngl) ? km : 0.0f;
                bOut[j] = br[nc] * s * nm;
            }
        }
    };

    v8f acc0 = {};
    v8f acc1 = {};
    float bR[8];

    issueA(0, 0);
    loadB(0, bR);

    int k0 = 0, p = 0;
    while (true) {
        // write B tile for this iteration (double-buffered: safe vs. iter-1)
#pragma unroll
        for (int j = 0; j < 4; ++j) {
            unsigned int pk = f2bf_pk(bR[2 * j], bR[2 * j + 1]);
            Blds[p][bn0 + 2 * j][bk]     = (unsigned short)pk;
            Blds[p][bn0 + 2 * j + 1][bk] = (unsigned short)(pk >> 16);
        }

        landA(p);              // A[p] slice complete (async: wait own slice)
        __syncthreads();       // whole A[p]/B[p] visible to all waves

        const int kn = k0 + BK;
        const bool more = kn < Kp;
        float bN[8];
        if (more) {
            issueA(kn, p ^ 1); // all waves passed barrier: A[p^1] readers done
            loadB(kn, bN);
        }

        v8u au;
        {
            const unsigned int* arow = (const unsigned int*)(&Alds[p][wm * 16 + l15][0]);
#pragma unroll
            for (int v = 0; v < 8; ++v) {
                int K0 = ((v >> 2) << 4) + (lhi << 3) + ((v & 3) << 1);
                au[v] = arow[K0 >> 1];
            }
        }
        v16bf afrag = __builtin_bit_cast(v16bf, au);

        v8u bu0, bu1;
        {
            const unsigned int* b0r = (const unsigned int*)(&Blds[p][wn2 * 32 + l15][0]);
            const unsigned int* b1r = (const unsigned int*)(&Blds[p][wn2 * 32 + 16 + l15][0]);
#pragma unroll
            for (int v = 0; v < 8; ++v) {
                int idx = ((v >> 2) << 3) + (lhi << 2) + (v & 3);
                bu0[v] = b0r[idx];
                bu1[v] = b1r[idx];
            }
        }
        v16bf bfrag0 = __builtin_bit_cast(v16bf, bu0);
        v16bf bfrag1 = __builtin_bit_cast(v16bf, bu1);

        acc0 = __builtin_amdgcn_wmma_f32_16x16x32_bf16(
            false, afrag, false, bfrag0, (short)0, acc0, false, false);
        acc1 = __builtin_amdgcn_wmma_f32_16x16x32_bf16(
            false, afrag, false, bfrag1, (short)0, acc1, false, false);

        if (!more) break;
#pragma unroll
        for (int j = 0; j < 8; ++j) bR[j] = bN[j];
        k0 = kn;
        p ^= 1;
    }

#pragma unroll
    for (int j = 0; j < 2; ++j) {
        v8f acc = j ? acc1 : acc0;
        long long n = colBase + wn2 * 32 + j * 16 + l15;
#pragma unroll
        for (int r = 0; r < 8; ++r) {
            long long m = rowBase + wm * 16 + lhi * 8 + r;
            if (m < M && n < Ngl) {
                float val = acc[r];
                if (rowscale) val *= rowscale[m];
                C[m * ldc + n] = val;
            }
        }
    }
}

// ---------------------------------------------------------------------------
// Host orchestration
// ---------------------------------------------------------------------------
static inline void launch_gemm(const float* A, long long lda,
                               const float* B, long long ldb,
                               float* C, long long ldc,
                               int M, int Ngl, int K,
                               const float* bRowScale,
                               const float* rowscale,
                               const float* alphaPtr,
                               const float* residual, long long ldr,
                               const float* cadd, long long ldadd,
                               const float* bias, int relu,
                               hipStream_t stream) {
    dim3 grid((Ngl + BN - 1) / BN, (M + BM - 1) / BM);
    gemm_bf16_kernel<<<grid, 256, 0, stream>>>(
        A, lda, B, ldb, C, ldc, M, Ngl, K,
        bRowScale, rowscale, alphaPtr, residual, ldr, cadd, ldadd, bias, relu);
}

static inline void launch_agg(const unsigned short* A, long long lda,
                              const float* B, long long ldb,
                              float* C, long long ldc,
                              int M, int Ngl, int K,
                              const float* bRowScale,
                              const float* rowscale,
                              hipStream_t stream) {
    dim3 grid((Ngl + BN - 1) / BN, (M + BM - 1) / BM);
    gemm_aggB_kernel<<<grid, 256, 0, stream>>>(
        A, lda, B, ldb, C, ldc, M, Ngl, K, bRowScale, rowscale);
}

extern "C" void kernel_launch(void* const* d_in, const int* in_sizes, int n_in,
                              void* d_out, int out_size, void* d_ws, size_t ws_size,
                              hipStream_t stream) {
    const float* x        = (const float*)d_in[0];
    const float* adj      = (const float*)d_in[1];
    const float* alpha    = (const float*)d_in[2];
    const float* rezero_w = (const float*)d_in[3];
    const float* w1       = (const float*)d_in[4];
    const float* rw1      = (const float*)d_in[5];
    const float* b1       = (const float*)d_in[6];
    const float* w2       = (const float*)d_in[7];
    const float* rw2      = (const float*)d_in[8];
    const float* b2       = (const float*)d_in[9];
    const float* w3       = (const float*)d_in[10];
    const float* rw3      = (const float*)d_in[11];
    const float* b3       = (const float*)d_in[12];

    const int IN_C = 128, HID = 256, OUT_C = 40;
    const int N = in_sizes[0] / IN_C;            // 10000

    // padded bf16 adjacency geometry
    const size_t MP = ((size_t)N + BM - 1) & ~(size_t)(BM - 1);
    const size_t KP = ((size_t)N + BK - 1) & ~(size_t)(BK - 1);
    const size_t adjB_bytes = MP * KP * sizeof(unsigned short);
    const size_t adjB_pad   = (adjB_bytes + 255) & ~(size_t)255;
    const size_t fbytes     = ((size_t)2 * N + (size_t)4 * N * HID) * sizeof(float);
    const bool useBf16Adj   = (ws_size >= adjB_pad + fbytes);

    unsigned short* adjB = (unsigned short*)d_ws;
    float* fbase = useBf16Adj ? (float*)((char*)d_ws + adjB_pad) : (float*)d_ws;

    float* dis   = fbase;
    float* scale = dis + N;
    float* agg   = scale + N;
    float* tmp   = agg + (size_t)N * HID;
    float* hA    = tmp + (size_t)N * HID;
    float* hB    = hA  + (size_t)N * HID;
    float* outp  = (float*)d_out;

    const float* NUL = nullptr;

    // Aggregation GEMM dispatcher: agg = rs_i * (adj @ (bs_j * Bmat))
    auto aggGemm = [&](const float* Bmat, int C, float* Cout,
                       const float* bs, const float* rs) {
        if (useBf16Adj)
            launch_agg(adjB, (long long)KP, Bmat, C, Cout, C, N, C, N, bs, rs, stream);
        else
            launch_gemm(adj, N, Bmat, C, Cout, C, N, C, N,
                        bs, rs, NUL, NUL, 0, NUL, 0, NUL, 0, stream);
    };

    if (useBf16Adj)
        stats_convert_kernel<<<(int)MP, 256, 0, stream>>>(adj, adjB, dis, scale, N, (int)KP);
    else
        stats_kernel<<<N, 256, 0, stream>>>(adj, dis, scale, N);

    // ---- rezero residual: hA = x + alpha * ((norm_adj @ x) @ rezero_w) ----
    aggGemm(x, IN_C, agg, dis, dis);                 // agg = norm_adj @ x
    launch_gemm(agg, IN_C, rezero_w, IN_C, hA, IN_C, N, IN_C, IN_C,
                NUL, NUL, alpha, x, IN_C, NUL, 0, NUL, 0, stream);

    // ---- layer 1: 128 -> 256, relu ----
    aggGemm(hA, IN_C, agg, dis, scale);              // mean aggregation
    launch_gemm(agg, IN_C, w1, HID, tmp, HID, N, HID, IN_C,
                NUL, NUL, NUL, NUL, 0, NUL, 0, NUL, 0, stream);
    launch_gemm(hA, IN_C, rw1, HID, hB, HID, N, HID, IN_C,
                NUL, NUL, NUL, NUL, 0, tmp, HID, b1, 1, stream);

    // ---- layer 2: 256 -> 256, relu ----
    aggGemm(hB, HID, agg, dis, scale);
    launch_gemm(agg, HID, w2, HID, tmp, HID, N, HID, HID,
                NUL, NUL, NUL, NUL, 0, NUL, 0, NUL, 0, stream);
    launch_gemm(hB, HID, rw2, HID, hA, HID, N, HID, HID,
                NUL, NUL, NUL, NUL, 0, tmp, HID, b2, 1, stream);

    // ---- layer 3: 256 -> 40, no relu ----
    aggGemm(hA, HID, agg, dis, scale);
    launch_gemm(agg, HID, w3, OUT_C, tmp, OUT_C, N, OUT_C, HID,
                NUL, NUL, NUL, NUL, 0, NUL, 0, NUL, 0, stream);
    launch_gemm(hA, HID, rw3, OUT_C, outp, OUT_C, N, OUT_C, HID,
                NUL, NUL, NUL, NUL, 0, tmp, OUT_C, b3, 0, stream);
}
